// FullAttention_30142080483989
// MI455X (gfx1250) — compile-verified, matrix operand order checked by reading, not run on hardware
//
#include <hip/hip_runtime.h>

// Flash attention forward, causal, B=2 H=16 L=2048 D=64, fp32 in/out.
// bf16 WMMA math on gfx1250 (wave32). bf16 staged in LDS in fragment-
// permuted layouts; each wave computes a 32-row Q tile (two M-tiles) so
// every K/V fragment feeds 2 WMMAs; row-sums come from a P x ones WMMA.
// Register-pressure-tuned to stay under 256 VGPRs/lane (no VGPR-MSB mode).

typedef __bf16 bf16_t;
typedef __attribute__((ext_vector_type(16))) __bf16 v16bf;
typedef __attribute__((ext_vector_type(8)))  __bf16 v8bf;
typedef __attribute__((ext_vector_type(8)))  float  v8f;
typedef __attribute__((ext_vector_type(4)))  float  f4;

#define BATCH 2
#define HEADS 16
#define SEQ   2048
#define DHEAD 64

#define BQ    128          // q rows per workgroup
#define NWAVE 4            // waves per workgroup (wave32), 32 q rows each
#define BK    32           // keys per inner chunk
#define KSTRB 72           // kbufB row stride (bf16)
#define VSTR  40           // vbufT row stride (bf16)
#define PSTRB 40           // pbufB row stride (bf16)

#define SCALE_LOG2E 0.180336880f   // (1/sqrt(64)) * log2(e), folded into Q

static __device__ __forceinline__ float fast_exp2(float x) {
  return __builtin_amdgcn_exp2f(x);                 // raw v_exp_f32
}

static __device__ __forceinline__ v16bf ld_frag(const bf16_t* p) {
  v8bf a = *(const v8bf*)p;
  v8bf b = *(const v8bf*)(p + 8);
  return __builtin_shufflevector(a, b, 0,1,2,3,4,5,6,7,8,9,10,11,12,13,14,15);
}

__global__ __launch_bounds__(128)
void fattn_causal_fwd(const float* __restrict__ Q,
                      const float* __restrict__ K,
                      const float* __restrict__ V,
                      float* __restrict__ O) {
  // K chunk: [key][perm(d)].  V chunk transposed: [d][perm(key)].
  // P scratch: per wave [row 0..31][perm(key-in-chunk)].
  __shared__ __align__(16) bf16_t kbufB[BK * KSTRB];
  __shared__ __align__(16) bf16_t vbufT[DHEAD * VSTR];
  __shared__ __align__(16) bf16_t pbufB[NWAVE * 32 * PSTRB];

  const int tid  = threadIdx.x;
  const int wave = tid >> 5;
  const int lane = tid & 31;
  const int hi   = lane >> 4;
  const int ln   = lane & 15;

  const int qb = blockIdx.x;
  const int bh = blockIdx.z * HEADS + blockIdx.y;
  const size_t headbase = (size_t)bh * SEQ * DHEAD;

  const int qrow0 = qb * BQ + wave * 32;            // first of this wave's 32 q rows

  // ---- Q A-fragments for two 16-row M-tiles, pre-scaled by 1/8*log2(e) ----
  v16bf qa[2][2];
  #pragma unroll
  for (int m = 0; m < 2; ++m) {
    const float* qp = Q + headbase + (size_t)(qrow0 + m * 16 + ln) * DHEAD;
    #pragma unroll
    for (int c = 0; c < 2; ++c) {
      const int d0 = c * 32 + hi * 8;
      #pragma unroll
      for (int e = 0; e < 8; ++e) {
        qa[m][c][e]     = (bf16_t)(qp[d0 + e]      * SCALE_LOG2E);
        qa[m][c][8 + e] = (bf16_t)(qp[d0 + 16 + e] * SCALE_LOG2E);
      }
    }
  }

  // Ones B-matrix for row-sum WMMA.
  v16bf ones;
  #pragma unroll
  for (int i = 0; i < 16; ++i) ones[i] = (bf16_t)1.0f;

  float m_run[2][8], l_run[2][8];
  #pragma unroll
  for (int m = 0; m < 2; ++m)
    #pragma unroll
    for (int e = 0; e < 8; ++e) { m_run[m][e] = -3.0e38f; l_run[m][e] = 0.0f; }
  v8f o[2][4] = { { {}, {}, {}, {} }, { {}, {}, {}, {} } };

  const int nkeys = (qb + 1) * BQ;
  const float* Kg = K + headbase;
  const float* Vg = V + headbase;

  // Staging: 128 threads move K/V chunk (32x64 fp32 each): 16 floats per thread.
  const int ldr = tid >> 2;                         // key row 0..31
  const int ldc = (tid & 3) * 16;                   // col 0..48 step 16
  int koff[2];                                      // permuted d-offsets
  #pragma unroll
  for (int h = 0; h < 2; ++h) {
    const int j  = (tid & 3) * 2 + h;               // 8-wide group 0..7
    const int jc = j >> 2, jj = j & 3;
    koff[h] = jc * 32 + (((jj & 1) << 1) | (jj >> 1)) * 8;
  }
  const int kj = ldr >> 3;
  const int pk = (ldr & 7) + 8 * (((kj & 1) << 1) | (kj >> 1)); // permuted key

  for (int kc = 0; kc < nkeys; kc += BK) {
    // ---- load + convert + store this chunk into LDS (no pipeline regs) ----
    {
      const float* kg = Kg + (size_t)(kc + ldr) * DHEAD + ldc;
      const float* vg = Vg + (size_t)(kc + ldr) * DHEAD + ldc;
      f4 kr0 = *(const f4*)(kg);      f4 kr1 = *(const f4*)(kg + 4);
      f4 kr2 = *(const f4*)(kg + 8);  f4 kr3 = *(const f4*)(kg + 12);
      f4 vr0 = *(const f4*)(vg);      f4 vr1 = *(const f4*)(vg + 4);
      f4 vr2 = *(const f4*)(vg + 8);  f4 vr3 = *(const f4*)(vg + 12);
      v8bf kba, kbb;
      #pragma unroll
      for (int i = 0; i < 4; ++i) {
        kba[i] = (bf16_t)kr0[i];  kba[4 + i] = (bf16_t)kr1[i];
        kbb[i] = (bf16_t)kr2[i];  kbb[4 + i] = (bf16_t)kr3[i];
      }
      *(v8bf*)&kbufB[ldr * KSTRB + koff[0]] = kba;
      *(v8bf*)&kbufB[ldr * KSTRB + koff[1]] = kbb;
      #pragma unroll
      for (int i = 0; i < 4; ++i) {
        vbufT[(ldc + i)      * VSTR + pk] = (bf16_t)vr0[i];
        vbufT[(ldc + 4 + i)  * VSTR + pk] = (bf16_t)vr1[i];
        vbufT[(ldc + 8 + i)  * VSTR + pk] = (bf16_t)vr2[i];
        vbufT[(ldc + 12 + i) * VSTR + pk] = (bf16_t)vr3[i];
      }
      // Register-free latency hiding: warm caches for the next two chunks.
      if (kc + BK < nkeys) {
        __builtin_prefetch(Kg + (size_t)(kc + BK + ldr) * DHEAD + ldc, 0, 3);
        __builtin_prefetch(Vg + (size_t)(kc + BK + ldr) * DHEAD + ldc, 0, 3);
      }
      if (kc + 2 * BK < nkeys) {
        __builtin_prefetch(Kg + (size_t)(kc + 2 * BK + ldr) * DHEAD + ldc, 0, 1);
        __builtin_prefetch(Vg + (size_t)(kc + 2 * BK + ldr) * DHEAD + ldc, 0, 1);
      }
    }
    __syncthreads();

    // ---- K B-fragments (shared by both M-tiles) ----
    v16bf kb0[2], kb1[2];
    #pragma unroll
    for (int t = 0; t < 2; ++t) {
      const bf16_t* krp = &kbufB[(t * 16 + ln) * KSTRB + hi * 16];
      kb0[t] = ld_frag(krp);
      kb1[t] = ld_frag(krp + 32);
    }

    // ---- per M-tile: scores -> mask -> online softmax -> P store ----
    const int pc0 = ln + 8 * (ln >> 3);             // perm col for tile 0
    #pragma unroll
    for (int m = 0; m < 2; ++m) {
      v8f s[2];
      #pragma unroll
      for (int t = 0; t < 2; ++t) {
        v8f z = {};
        v8f acc = __builtin_amdgcn_wmma_f32_16x16x32_bf16(false, qa[m][0], false, kb0[t],
                                                          (short)0, z, false, false);
        s[t]    = __builtin_amdgcn_wmma_f32_16x16x32_bf16(false, qa[m][1], false, kb1[t],
                                                          (short)0, acc, false, false);
      }
      const int mrow0 = qrow0 + m * 16;
      const bool needmask = (kc + BK - 1) > mrow0;
      float mc[8];
      #pragma unroll
      for (int e = 0; e < 8; ++e) {
        const int rg = mrow0 + e + hi * 8;
        float s0 = s[0][e];
        float s1 = s[1][e];
        if (needmask) {
          if (kc + ln      > rg) s0 = -1.0e30f;
          if (kc + 16 + ln > rg) s1 = -1.0e30f;
        }
        s[0][e] = s0; s[1][e] = s1;
        mc[e] = fmaxf(s0, s1);
      }
      #pragma unroll
      for (int e = 0; e < 8; ++e) {                 // row max across 16 cols
        mc[e] = fmaxf(mc[e], __shfl_xor(mc[e], 1, 32));
        mc[e] = fmaxf(mc[e], __shfl_xor(mc[e], 2, 32));
        mc[e] = fmaxf(mc[e], __shfl_xor(mc[e], 4, 32));
        mc[e] = fmaxf(mc[e], __shfl_xor(mc[e], 8, 32));
      }
      bf16_t* pw = &pbufB[(wave * 32 + m * 16) * PSTRB];
      #pragma unroll
      for (int e = 0; e < 8; ++e) {
        const float mn = fmaxf(m_run[m][e], mc[e]);
        const float cr = fast_exp2(m_run[m][e] - mn);
        m_run[m][e] = mn;
        l_run[m][e] *= cr;
        const float p0 = fast_exp2(s[0][e] - mn);
        const float p1 = fast_exp2(s[1][e] - mn);
        pw[(e + hi * 8) * PSTRB + pc0]     = (bf16_t)p0;
        pw[(e + hi * 8) * PSTRB + pc0 + 8] = (bf16_t)p1;
        #pragma unroll
        for (int t = 0; t < 4; ++t)
          o[m][t][e] *= cr;
      }
    }

    // Per-wave P transpose through LDS: same-wave LDS ops are in-order on
    // CDNA5; only a compiler scheduling barrier is needed.
    __builtin_amdgcn_wave_barrier();

    // ---- V B-fragments, then row-sum WMMA + 8 PV WMMAs ----
    v16bf vb[4];
    #pragma unroll
    for (int t = 0; t < 4; ++t)
      vb[t] = ld_frag(&vbufT[(t * 16 + ln) * VSTR + hi * 16]);

    #pragma unroll
    for (int m = 0; m < 2; ++m) {
      v16bf pa = ld_frag(&pbufB[(wave * 32 + m * 16 + ln) * PSTRB + hi * 16]);
      v8f z = {};
      v8f rst = __builtin_amdgcn_wmma_f32_16x16x32_bf16(false, pa, false, ones,
                                                        (short)0, z, false, false);
      #pragma unroll
      for (int e = 0; e < 8; ++e) l_run[m][e] += rst[e];
      #pragma unroll
      for (int t = 0; t < 4; ++t)
        o[m][t] = __builtin_amdgcn_wmma_f32_16x16x32_bf16(false, pa, false, vb[t],
                                                          (short)0, o[m][t], false, false);
    }

    __syncthreads();   // all waves done with kbufB/vbufT before next staging
  }

  // ---- normalize and write out ----
  #pragma unroll
  for (int m = 0; m < 2; ++m) {
    float inv[8];
    #pragma unroll
    for (int e = 0; e < 8; ++e) inv[e] = 1.0f / l_run[m][e];
    #pragma unroll
    for (int t = 0; t < 4; ++t)
      #pragma unroll
      for (int e = 0; e < 8; ++e) {
        const size_t idx = headbase +
            (size_t)(qrow0 + m * 16 + e + hi * 8) * DHEAD + t * 16 + ln;
        O[idx] = o[m][t][e] * inv[e];
      }
  }
}

extern "C" void kernel_launch(void* const* d_in, const int* in_sizes, int n_in,
                              void* d_out, int out_size, void* d_ws, size_t ws_size,
                              hipStream_t stream) {
  (void)in_sizes; (void)n_in; (void)out_size; (void)d_ws; (void)ws_size;
  const float* Q = (const float*)d_in[0];
  const float* K = (const float*)d_in[1];
  const float* V = (const float*)d_in[2];
  // d_in[3] is the boolean causal mask; causality is computed analytically.
  float* O = (float*)d_out;

  dim3 grid(SEQ / BQ, HEADS, BATCH);
  dim3 block(128);
  hipLaunchKernelGGL(fattn_causal_fwd, grid, block, 0, stream, Q, K, V, O);
}
